// Eva_10866267259233
// MI455X (gfx1250) — matrix operand order repaired and checked by reading, not compiled, not run on hardware
//
#include <hip/hip_runtime.h>

#define BATCH 4
#define SEQ   1024
#define DIMC  1024
#define NH    16
#define DH    64
#define GK    1024   // K for both GEMMs (DIMC)

typedef __attribute__((ext_vector_type(16))) __bf16          v16bf;
typedef __attribute__((ext_vector_type(16))) unsigned short  v16u;
typedef __attribute__((ext_vector_type(8)))  unsigned short  v8u;
typedef __attribute__((ext_vector_type(8)))  float           v8f;

__device__ __forceinline__ unsigned short f2bf(float f) {
    unsigned u = __builtin_bit_cast(unsigned, f);
    unsigned r = (u + 0x7FFFu + ((u >> 16) & 1u)) >> 16;
    return (unsigned short)r;
}
__device__ __forceinline__ float bf2f(unsigned short h) {
    unsigned u = ((unsigned)h) << 16;
    return __builtin_bit_cast(float, u);
}
__device__ __forceinline__ v16bf load_frag16(const unsigned short* p0,
                                             const unsigned short* p1) {
    union { v16u v; v8u h[2]; } u;
    u.h[0] = *(const v8u*)p0;
    u.h[1] = *(const v8u*)p1;
    return __builtin_bit_cast(v16bf, u.v);
}

// ---------------- elementwise converts ----------------
__global__ void cvt_bf16_kernel(const float* __restrict__ in,
                                unsigned short* __restrict__ out, int n) {
    int i = blockIdx.x * blockDim.x + threadIdx.x;
    if (i < n) out[i] = f2bf(in[i]);
}

// in: rows x cols row-major; out: cols x rows row-major (i.e. transposed)
__global__ void cvt_transpose_kernel(const float* __restrict__ in,
                                     unsigned short* __restrict__ out,
                                     int rows, int cols) {
    int i = blockIdx.x * blockDim.x + threadIdx.x;
    if (i < rows * cols) {
        int c = i / rows;      // output row  (= input col)
        int r = i - c * rows;  // output col  (= input row)
        out[i] = f2bf(in[(size_t)r * cols + c]);
    }
}

// vb (B,N,H,D) bf16 -> vT (B,H,D,N) bf16
__global__ void transpose_v_kernel(const unsigned short* __restrict__ vb,
                                   unsigned short* __restrict__ vT) {
    int i = blockIdx.x * blockDim.x + threadIdx.x;
    if (i < BATCH * NH * DH * SEQ) {
        int n  = i & (SEQ - 1);
        int d  = (i >> 10) & (DH - 1);
        int hh = (i >> 16) & (NH - 1);
        int bb = i >> 20;
        vT[i] = vb[((size_t)(bb * SEQ + n)) * DIMC + hh * DH + d];
    }
}

// ---------------- RoPE (in place on bf16 q or k) ----------------
__global__ void rope_kernel(unsigned short* __restrict__ q,
                            const float* __restrict__ cs,
                            const float* __restrict__ sn) {
    int t = blockIdx.x * blockDim.x + threadIdx.x;   // B*N*H*32 threads
    if (t >= BATCH * SEQ * NH * 32) return;
    int d  = t & 31;
    int hh = (t >> 5) & (NH - 1);
    int n  = (t >> 9) & (SEQ - 1);
    int bb = t >> 19;
    size_t base = (size_t)(bb * SEQ + n) * DIMC + hh * DH;
    float q0 = bf2f(q[base + d]);
    float q1 = bf2f(q[base + d + 32]);
    float c0 = cs[n * DH + d],      s0 = sn[n * DH + d];
    float c1 = cs[n * DH + d + 32], s1 = sn[n * DH + d + 32];
    q[base + d]      = f2bf(q0 * c0 - q1 * s0);
    q[base + d + 32] = f2bf(q1 * c1 + q0 * s1);
}

// ---------------- g = log_sigmoid(x@Wg + bg), G[b,h,n] ----------------
__global__ void __launch_bounds__(128)
gate_kernel(const float* __restrict__ x, const float* __restrict__ Wg,
            const float* __restrict__ bg, float* __restrict__ G) {
    __shared__ float xs[DIMC];
    __shared__ float ps[128];
    int bid = blockIdx.x;            // b*SEQ + n
    int bb = bid >> 10;
    int n  = bid & (SEQ - 1);
    int tid = threadIdx.x;
    for (int i = tid; i < DIMC; i += 128)
        xs[i] = x[(size_t)(bb * SEQ + n) * DIMC + i];
    __syncthreads();
    int h = tid & 15;
    int chunk = tid >> 4;            // 8 chunks of 128
    float p = 0.f;
    int k0 = chunk * 128;
    for (int k = k0; k < k0 + 128; ++k) p += xs[k] * Wg[(size_t)k * NH + h];
    ps[tid] = p;
    __syncthreads();
    if (tid < NH) {
        float z = bg[tid];
        for (int c = 0; c < 8; ++c) z += ps[c * 16 + tid];
        float m = fminf(z, 0.f);
        float v = m - log1pf(__expf(-fabsf(z)));   // stable log_sigmoid
        G[(size_t)(bb * NH + tid) * SEQ + n] = v;
    }
}

// ---------------- cumsum over n per (b,h), one wave each ----------------
__global__ void cumsum_kernel(float* __restrict__ G) {
    int base = blockIdx.x * SEQ;     // 64 blocks of 32 threads
    int lane = threadIdx.x;
    float off = 0.f;
    for (int c = 0; c < SEQ; c += 32) {
        float v = G[base + c + lane];
        #pragma unroll
        for (int s = 1; s < 32; s <<= 1) {
            float t = __shfl_up(v, s, 32);
            if (lane >= s) v += t;
        }
        v += off;
        G[base + c + lane] = v;
        off = __shfl(v, 31, 32);
    }
}

// ---------------- generic bf16 WMMA GEMM: C(MxN)=A(MxK)@W(KxN)+bias ------
// BT is W transposed (N x K) so B-fragments load contiguously along K.
// All 8 waves of a block share one 16-row A slab -> async-staged to LDS.
// mode 0: write bf16 to outb ; mode 1: write f32 to outf
__global__ void __launch_bounds__(256)
gemm_bf16_kernel(const unsigned short* __restrict__ A,
                 const unsigned short* __restrict__ BT,
                 const float* __restrict__ bias,
                 unsigned short* __restrict__ outb,
                 float* __restrict__ outf,
                 int M, int N, int mode) {
    __shared__ __align__(16) unsigned short As[16 * GK];   // 32 KB A slab
    const int tid  = threadIdx.x;
    const int lane = tid & 31;
    const int wid  = tid >> 5;                 // 0..7
    const int tiles_n = N >> 6;                // 16
    const int blocks_per_tm = tiles_n >> 3;    // 2
    const int tm = blockIdx.x / blocks_per_tm;
    const int tn = (blockIdx.x % blocks_per_tm) * 8 + wid;
    const int m0 = tm << 4, n0 = tn << 6;
    const int hv  = (lane >> 4) & 1;
    const int l15 = lane & 15;

    // ---- cooperative async stage of A slab (16 x GK bf16) into LDS ----
    // 2048 x 16B transfers; each lane moves 16B per op, 8 ops per thread.
    #pragma unroll
    for (int c = 0; c < (GK >> 7); ++c) {          // 8
        int off = (tid + c * 256) * 8;             // element (half) offset
        int row = off >> 10;                       // / GK
        int kk  = off & (GK - 1);
        unsigned lds_addr = (unsigned)(uintptr_t)(As + off);
        const unsigned short* g = A + (size_t)(m0 + row) * GK + kk;
        asm volatile("global_load_async_to_lds_b128 %0, %1, off"
                     :: "v"(lds_addr), "v"(g) : "memory");
    }
    asm volatile("s_wait_asynccnt 0x0" ::: "memory");
    __syncthreads();

    const unsigned short* a0 = As + (size_t)l15 * GK + hv * 16;
    const unsigned short* b0 = BT + (size_t)(n0 +  0 + l15) * GK + hv * 16;
    const unsigned short* b1 = BT + (size_t)(n0 + 16 + l15) * GK + hv * 16;
    const unsigned short* b2 = BT + (size_t)(n0 + 32 + l15) * GK + hv * 16;
    const unsigned short* b3 = BT + (size_t)(n0 + 48 + l15) * GK + hv * 16;

    v8f c0 = {}, c1 = {}, c2 = {}, c3 = {};
    for (int k0 = 0; k0 < GK; k0 += 32) {
        v16bf a  = load_frag16(a0 + k0, a0 + k0 + 8);
        v16bf f0 = load_frag16(b0 + k0, b0 + k0 + 8);
        v16bf f1 = load_frag16(b1 + k0, b1 + k0 + 8);
        v16bf f2 = load_frag16(b2 + k0, b2 + k0 + 8);
        v16bf f3 = load_frag16(b3 + k0, b3 + k0 + 8);
        c0 = __builtin_amdgcn_wmma_f32_16x16x32_bf16(false, a, false, f0, (short)0, c0, false, false);
        c1 = __builtin_amdgcn_wmma_f32_16x16x32_bf16(false, a, false, f1, (short)0, c1, false, false);
        c2 = __builtin_amdgcn_wmma_f32_16x16x32_bf16(false, a, false, f2, (short)0, c2, false, false);
        c3 = __builtin_amdgcn_wmma_f32_16x16x32_bf16(false, a, false, f3, (short)0, c3, false, false);
    }
    float bi0 = bias[n0 +  0 + l15];
    float bi1 = bias[n0 + 16 + l15];
    float bi2 = bias[n0 + 32 + l15];
    float bi3 = bias[n0 + 48 + l15];
    #pragma unroll
    for (int r = 0; r < 8; ++r) {
        size_t ro = (size_t)(m0 + r + hv * 8) * N + n0 + l15;
        if (mode == 0) {
            outb[ro +  0] = f2bf(c0[r] + bi0);
            outb[ro + 16] = f2bf(c1[r] + bi1);
            outb[ro + 32] = f2bf(c2[r] + bi2);
            outb[ro + 48] = f2bf(c3[r] + bi3);
        } else {
            outf[ro +  0] = c0[r] + bi0;
            outf[ro + 16] = c1[r] + bi1;
            outf[ro + 32] = c2[r] + bi2;
            outf[ro + 48] = c3[r] + bi3;
        }
    }
}

// ---------------- attention: one wave per (b,h,16-row i-tile) ------------
__global__ void __launch_bounds__(128)
attn_kernel(const unsigned short* __restrict__ qb,
            const unsigned short* __restrict__ kb,
            const unsigned short* __restrict__ vT,
            const float* __restrict__ G,
            unsigned short* __restrict__ ob) {
    __shared__ __align__(16) unsigned short lds_w[4][16 * 32]; // per-wave w tile
    const int lane  = threadIdx.x & 31;
    const int wslot = (threadIdx.x >> 5) & 3;
    const int w  = (blockIdx.x * blockDim.x + threadIdx.x) >> 5;
    const int it = w & 63;
    const int h  = (w >> 6) & (NH - 1);
    const int b  = w >> 10;
    const int i0 = it << 4;
    const int hv  = (lane >> 4) & 1;
    const int l15 = lane & 15;
    unsigned short* wt = &lds_w[wslot][0];

    // q A-fragments (16x32, two K-chunks covering D=64)
    const unsigned short* qrow = qb + (size_t)(b * SEQ + i0 + l15) * DIMC + h * DH;
    v16bf aq0 = load_frag16(qrow + hv * 16,      qrow + hv * 16 + 8);
    v16bf aq1 = load_frag16(qrow + 32 + hv * 16, qrow + 32 + hv * 16 + 8);

    const float* Gh = G + (size_t)(b * NH + h) * SEQ;
    float Gi[8];
    #pragma unroll
    for (int r = 0; r < 8; ++r) Gi[r] = Gh[i0 + r + hv * 8];

    v8f acc0 = {}, acc1 = {}, acc2 = {}, acc3 = {};
    float rs[8] = {0, 0, 0, 0, 0, 0, 0, 0};

    const unsigned short* vhead = vT + (size_t)((b * NH + h) * DH) * SEQ;

    for (int jp = 0; jp <= it; jp += 2) {
        #pragma unroll
        for (int u = 0; u < 2; ++u) {
            const int jt = jp + u;            // wave-uniform
            if (jt <= it) {
                const int j0u = jt << 4;
                const unsigned short* krow =
                    kb + (size_t)(b * SEQ + j0u + l15) * DIMC + h * DH;
                v16bf bk0 = load_frag16(krow + hv * 16,      krow + hv * 16 + 8);
                v16bf bk1 = load_frag16(krow + 32 + hv * 16, krow + 32 + hv * 16 + 8);
                v8f s = {};
                s = __builtin_amdgcn_wmma_f32_16x16x32_bf16(false, aq0, false, bk0, (short)0, s, false, false);
                s = __builtin_amdgcn_wmma_f32_16x16x32_bf16(false, aq1, false, bk1, (short)0, s, false, false);
                const int j = j0u + l15;
                const float Gj = Gh[j];
                #pragma unroll
                for (int r = 0; r < 8; ++r) {
                    const int i = i0 + r + hv * 8;
                    float val = 0.0f;
                    if (j <= i) {
                        float sv = s[r] * 0.125f;          // SCALE = D^-0.5
                        val = sv * sv * __expf(Gi[r] - Gj);
                    }
                    rs[r] += val;
                    wt[(r + hv * 8) * 32 + u * 16 + l15] = f2bf(val);
                }
            } else {
                #pragma unroll
                for (int r = 0; r < 8; ++r)
                    wt[(r + hv * 8) * 32 + 16 + l15] = 0;
            }
        }
        __builtin_amdgcn_wave_barrier();
        asm volatile("s_wait_dscnt 0x0" ::: "memory");   // LDS bounce visible
        v16bf aw = load_frag16(wt + l15 * 32 + hv * 16,
                               wt + l15 * 32 + hv * 16 + 8);
        const int jbase = (jp << 4) + hv * 16;
        const unsigned short* v0 = vhead + (size_t)( 0 + l15) * SEQ + jbase;
        const unsigned short* v1 = vhead + (size_t)(16 + l15) * SEQ + jbase;
        const unsigned short* v2 = vhead + (size_t)(32 + l15) * SEQ + jbase;
        const unsigned short* v3 = vhead + (size_t)(48 + l15) * SEQ + jbase;
        v16bf bv0 = load_frag16(v0, v0 + 8);
        v16bf bv1 = load_frag16(v1, v1 + 8);
        v16bf bv2 = load_frag16(v2, v2 + 8);
        v16bf bv3 = load_frag16(v3, v3 + 8);
        acc0 = __builtin_amdgcn_wmma_f32_16x16x32_bf16(false, aw, false, bv0, (short)0, acc0, false, false);
        acc1 = __builtin_amdgcn_wmma_f32_16x16x32_bf16(false, aw, false, bv1, (short)0, acc1, false, false);
        acc2 = __builtin_amdgcn_wmma_f32_16x16x32_bf16(false, aw, false, bv2, (short)0, acc2, false, false);
        acc3 = __builtin_amdgcn_wmma_f32_16x16x32_bf16(false, aw, false, bv3, (short)0, acc3, false, false);
        __builtin_amdgcn_wave_barrier();
    }

    // reduce row sums across the 16 lanes of each half, invert
    #pragma unroll
    for (int r = 0; r < 8; ++r) {
        float v = rs[r];
        v += __shfl_xor(v, 1, 32);
        v += __shfl_xor(v, 2, 32);
        v += __shfl_xor(v, 4, 32);
        v += __shfl_xor(v, 8, 32);
        rs[r] = 1.0f / (v + 1e-6f);
    }
    #pragma unroll
    for (int r = 0; r < 8; ++r) {
        size_t ro = (size_t)(b * SEQ + i0 + r + hv * 8) * DIMC + h * DH + l15;
        ob[ro +  0] = f2bf(acc0[r] * rs[r]);
        ob[ro + 16] = f2bf(acc1[r] * rs[r]);
        ob[ro + 32] = f2bf(acc2[r] * rs[r]);
        ob[ro + 48] = f2bf(acc3[r] * rs[r]);
    }
}

// ---------------- host launch ----------------
extern "C" void kernel_launch(void* const* d_in, const int* in_sizes, int n_in,
                              void* d_out, int out_size, void* d_ws, size_t ws_size,
                              hipStream_t stream) {
    (void)in_sizes; (void)n_in; (void)out_size; (void)ws_size;
    const float* x    = (const float*)d_in[0];
    const float* cs   = (const float*)d_in[1];
    const float* sn   = (const float*)d_in[2];
    const float* Wq   = (const float*)d_in[3];
    const float* bq   = (const float*)d_in[4];
    const float* Wk   = (const float*)d_in[5];
    const float* bk   = (const float*)d_in[6];
    const float* Wv   = (const float*)d_in[7];
    const float* bv   = (const float*)d_in[8];
    const float* Wg   = (const float*)d_in[9];
    const float* bg   = (const float*)d_in[10];
    const float* Wp   = (const float*)d_in[11];
    const float* bp   = (const float*)d_in[12];
    float* out = (float*)d_out;

    const size_t NTOT = (size_t)BATCH * SEQ * DIMC;       // 4,194,304
    const size_t WSZ  = (size_t)DIMC * DIMC;              // 1,048,576
    char* ws = (char*)d_ws;
    size_t off = 0;
    unsigned short* xb  = (unsigned short*)(ws + off); off += NTOT * 2;
    unsigned short* WqT = (unsigned short*)(ws + off); off += WSZ * 2;
    unsigned short* WkT = (unsigned short*)(ws + off); off += WSZ * 2;
    unsigned short* WvT = (unsigned short*)(ws + off); off += WSZ * 2;
    unsigned short* WpT = (unsigned short*)(ws + off); off += WSZ * 2;
    unsigned short* qbf = (unsigned short*)(ws + off); off += NTOT * 2;
    unsigned short* kbf = (unsigned short*)(ws + off); off += NTOT * 2;
    unsigned short* vbf = (unsigned short*)(ws + off); off += NTOT * 2;
    unsigned short* vT  = (unsigned short*)(ws + off); off += NTOT * 2;
    float*          Gb  = (float*)(ws + off);          off += (size_t)BATCH * NH * SEQ * 4;
    unsigned short* ob  = (unsigned short*)(ws + off); off += NTOT * 2;

    // 1) convert inputs
    cvt_bf16_kernel<<<(NTOT + 255) / 256, 256, 0, stream>>>(x, xb, (int)NTOT);
    cvt_transpose_kernel<<<(WSZ + 255) / 256, 256, 0, stream>>>(Wq, WqT, DIMC, DIMC);
    cvt_transpose_kernel<<<(WSZ + 255) / 256, 256, 0, stream>>>(Wk, WkT, DIMC, DIMC);
    cvt_transpose_kernel<<<(WSZ + 255) / 256, 256, 0, stream>>>(Wv, WvT, DIMC, DIMC);
    cvt_transpose_kernel<<<(WSZ + 255) / 256, 256, 0, stream>>>(Wp, WpT, DIMC, DIMC);

    // 2) q/k/v projections (WMMA GEMM): M=4096, N=1024, K=1024
    const int M = BATCH * SEQ;
    const int gemm_blocks = (M / 16) * ((DIMC / 64) / 8);          // 512
    gemm_bf16_kernel<<<gemm_blocks, 256, 0, stream>>>(xb, WqT, bq, qbf, nullptr, M, DIMC, 0);
    gemm_bf16_kernel<<<gemm_blocks, 256, 0, stream>>>(xb, WkT, bk, kbf, nullptr, M, DIMC, 0);
    gemm_bf16_kernel<<<gemm_blocks, 256, 0, stream>>>(xb, WvT, bv, vbf, nullptr, M, DIMC, 0);

    // 3) RoPE on q and k
    const int rope_threads = BATCH * SEQ * NH * 32;
    rope_kernel<<<rope_threads / 256, 256, 0, stream>>>(qbf, cs, sn);
    rope_kernel<<<rope_threads / 256, 256, 0, stream>>>(kbf, cs, sn);

    // 4) gate + cumsum -> G[b,h,n]
    gate_kernel<<<BATCH * SEQ, 128, 0, stream>>>(x, Wg, bg, Gb);
    cumsum_kernel<<<BATCH * NH, 32, 0, stream>>>(Gb);

    // 5) v transpose per head
    transpose_v_kernel<<<(NTOT + 255) / 256, 256, 0, stream>>>(vbf, vT);

    // 6) attention: B*H*(N/16) = 4096 waves, 4 waves / block
    attn_kernel<<<(BATCH * NH * (SEQ / 16)) / 4, 128, 0, stream>>>(qbf, kbf, vT, Gb, ob);

    // 7) output projection -> f32 d_out
    gemm_bf16_kernel<<<gemm_blocks, 256, 0, stream>>>(ob, WpT, bp, nullptr, out, M, DIMC, 1);
}
